// Model35_65962107732445
// MI455X (gfx1250) — matrix operand order repaired and checked by reading, not compiled, not run on hardware
//
#include <hip/hip_runtime.h>
#include <hip/hip_bf16.h>
#include <math.h>

typedef __attribute__((ext_vector_type(16))) _Float16 v16h;
typedef __attribute__((ext_vector_type(8)))  float    v8f;
typedef unsigned int u32x4 __attribute__((ext_vector_type(4)));
typedef int          i32x4 __attribute__((ext_vector_type(4)));
typedef int          i32x8 __attribute__((ext_vector_type(8)));

#define E      30
#define EP     32       // padded feature dim (WMMA K)
#define HD     10       // head dim
#define WAVES  4        // waves per block
#define NITER  10       // tiles per wave
#define TILE_ELEMS 8    // 16 rows = 8 elems x 2 tokens

#if defined(__has_builtin)
#  if __has_builtin(__builtin_amdgcn_tensor_load_to_lds)
#    define HAVE_TDM 1
#  endif
#  if __has_builtin(__builtin_amdgcn_s_wait_tensorcnt)
#    define HAVE_TENSORCNT_BUILTIN 1
#  endif
#endif

__device__ __forceinline__ void wait_tensorcnt0() {
#if defined(HAVE_TENSORCNT_BUILTIN)
  __builtin_amdgcn_s_wait_tensorcnt(0);
#else
  asm volatile("s_wait_tensorcnt 0" ::: "memory");
#endif
  asm volatile("" ::: "memory");
}

// Wave-local LDS fence: same-wave DS ops are in-order (CDNA5 ISA 7.3), so a
// dscnt drain + compiler memory barrier makes intra-wave cross-lane LDS
// communication safe without block-wide s_barrier.
__device__ __forceinline__ void wave_lds_fence() {
  asm volatile("s_wait_dscnt 0" ::: "memory");
}

__device__ __forceinline__ float gelu_exact(float x) {
  return 0.5f * x * (1.0f + erff(x * 0.70710678118654752f));
}

// A-fragment (16x32 f16) gather from an LDS 16x32 row-major fp32 buffer.
// lanes 0-15: VGPR0..3 -> K=0..7, VGPR4..7 -> K=16..23 (M = lane)
// lanes 16-31: K=8..15 and K=24..31.
__device__ __forceinline__ v16h load_a_frag(const float* buf, int lane) {
  const int m = lane & 15;
  const int h = lane >> 4;
  v16h a;
#pragma unroll
  for (int i = 0; i < 16; ++i) {
    const int p = i >> 1, s = i & 1;
    const int k = (p >= 4 ? 16 : 0) + h * 8 + ((p & 3) << 1) + s;
    a[i] = (_Float16)buf[m * EP + k];
  }
  return a;
}

// B-fragment (32x16 f16) from row-major weight W[outF][E] (einsum 'bsd,fd').
// lane n = L&15 selects column; lanes 0-15 hold K=0..7,16..23, lanes 16-31 K=8..15,24..31.
__device__ __forceinline__ v16h load_b_frag(const float* __restrict__ W,
                                            int outF, int tile, int lane) {
  const int n = lane & 15;
  const int h = lane >> 4;
  const int col = tile * 16 + n;
  v16h b;
#pragma unroll
  for (int i = 0; i < 16; ++i) {
    const int p = i >> 1, s = i & 1;
    const int k = (p >= 4 ? 16 : 0) + h * 8 + ((p & 3) << 1) + s;
    float v = 0.0f;
    if (k < E && col < outF) v = W[col * E + k];
    b[i] = (_Float16)v;
  }
  return b;
}

__device__ __forceinline__ v8f wmma16(v16h a, v16h b) {
  v8f c = {};
  return __builtin_amdgcn_wmma_f32_16x16x32_f16(false, a, false, b,
                                                (short)0, c, false, false);
}

// Full-wave layernorm over 16 rows x 30 features: lane pairs (l, l+16) share
// row l&15; each half-wave reduces 15 features, combined with shfl_xor(16).
__device__ __forceinline__ void layernorm16(float (*X)[EP], int lane,
                                            const float* __restrict__ gamma,
                                            const float* __restrict__ beta) {
  const int r    = lane & 15;
  const int base = (lane >> 4) * 15;   // 0 or 15
  float s = 0.f, s2 = 0.f;
#pragma unroll
  for (int i = 0; i < 15; ++i) {
    const float v = X[r][base + i];
    s += v; s2 += v * v;
  }
  s  += __shfl_xor(s, 16);
  s2 += __shfl_xor(s2, 16);
  const float mu = s * (1.0f / E);
  const float rs = rsqrtf(s2 * (1.0f / E) - mu * mu + 1e-5f);
#pragma unroll
  for (int i = 0; i < 15; ++i) {
    const int e = base + i;
    X[r][e] = (X[r][e] - mu) * rs * gamma[e] + beta[e];
  }
}

__global__ __launch_bounds__(WAVES * 32)
void tiny_transformer_wmma(
    const float* __restrict__ in,     // [B,2,5]
    const float* __restrict__ w_emb,  // [30,5]
    const float* __restrict__ b_emb,  // [30]
    const float* __restrict__ w_in,   // [90,30]
    const float* __restrict__ b_in,   // [90]
    const float* __restrict__ w_out,  // [30,30]
    const float* __restrict__ b_out,  // [30]
    const float* __restrict__ w1,     // [30,30]
    const float* __restrict__ b1,     // [30]
    const float* __restrict__ w2,     // [30,30]
    const float* __restrict__ b2,     // [30]
    const float* __restrict__ gamma,  // [30]
    const float* __restrict__ beta,   // [30]
    const float* __restrict__ w_o,    // [3,30]
    const float* __restrict__ b_o,    // [3]
    float* __restrict__ out,          // [B,3]
    int B)
{
  __shared__ float xb[WAVES][16][EP];              // residual stream
  __shared__ float qb[WAVES][16][96];              // qkv
  __shared__ float cb[WAVES][16][EP];              // ctx / ffn hidden
  __shared__ __align__(16) float ib[WAVES][80];    // staged input tile

  const int lane = threadIdx.x & 31;
  const int wave = threadIdx.x >> 5;
  float (*X)[EP] = xb[wave];
  float (*Q)[96] = qb[wave];
  float (*C)[EP] = cb[wave];
  float* IB      = ib[wave];

  // ---- wave-resident weight fragments (reused NITER times) ----
  v16h Bqkv[6], Bo[2], B1f[2], B2f[2];
#pragma unroll
  for (int t = 0; t < 6; ++t) Bqkv[t] = load_b_frag(w_in, 90, t, lane);
#pragma unroll
  for (int t = 0; t < 2; ++t) {
    Bo[t]  = load_b_frag(w_out, 30, t, lane);
    B1f[t] = load_b_frag(w1,   30, t, lane);
    B2f[t] = load_b_frag(w2,   30, t, lane);
  }

  // per-lane embed weights (lane = output feature) and scatter biases
  float we[5], be;
  {
    const int e = (lane < E) ? lane : 0;
#pragma unroll
    for (int j = 0; j < 5; ++j) we[j] = w_emb[e * 5 + j];
    be = b_emb[e];
  }
  const int n16 = lane & 15;
  const int hh  = lane >> 4;
  float bqkv[6], bo2[2], bb1[2], bb2[2];
#pragma unroll
  for (int t = 0; t < 6; ++t) {
    const int col = t * 16 + n16;
    bqkv[t] = (col < 90) ? b_in[col] : 0.f;
  }
#pragma unroll
  for (int t = 0; t < 2; ++t) {
    const int col = t * 16 + n16;
    bo2[t] = (col < E) ? b_out[col] : 0.f;
    bb1[t] = (col < E) ? b1[col]    : 0.f;
    bb2[t] = (col < E) ? b2[col]    : 0.f;
  }

#if defined(HAVE_TDM)
  // Static part of the TDM D# (group1): 1-D tile, data_size=4B,
  // tensor_dim0 = tile_dim0 = 80 elems, dim1 = 1, stride = 80.
  i32x8 g1;
  g1[0] = (int)(2u << 16);      // data_size=2 (4 bytes); workgroup_mask=0
  g1[1] = (int)(80u << 16);     // tensor_dim0[15:0] at bits 63:48
  g1[2] = (int)(1u << 16);      // tensor_dim0 hi = 0; tensor_dim1 = 1
  g1[3] = (int)(80u << 16);     // tensor_dim1 hi = 0; tile_dim0 = 80
  g1[4] = 1;                    // tile_dim1 = 1, tile_dim2 = 0
  g1[5] = 80;                   // tensor_dim0_stride low 32
  g1[6] = 0;
  g1[7] = 0;
  const i32x4 z4 = {0, 0, 0, 0};
  const unsigned int ldsIB = (unsigned int)(uintptr_t)IB;  // LDS = flat[31:0]
#endif

  for (int it = 0; it < NITER; ++it) {
    const long elemBase =
        ((long)(blockIdx.x * NITER + it) * WAVES + wave) * TILE_ELEMS;

    // prefetch next tile's input lines (-> global_prefetch_b8)
    if (it + 1 < NITER && lane < 20) {
      __builtin_prefetch(in + (elemBase + 32) * 10 + lane * 16, 0, 1);
    }

    // ---- stage input tile to LDS: 16 rows x 5 floats = 320 bytes ----
    bool staged = false;
#if defined(HAVE_TDM)
    if (elemBase + TILE_ELEMS <= (long)B) {
      const unsigned long long ga =
          (unsigned long long)(uintptr_t)(in + elemBase * 10);
      u32x4 g0;
      g0[0] = 1u;                                   // count=1 (valid D#)
      g0[1] = ldsIB;                                // lds_addr
      g0[2] = (unsigned int)ga;                     // global_addr[31:0]
      g0[3] = (unsigned int)((ga >> 32) & 0x1FFFFFFu) | (2u << 30); // type=2
#if __clang_major__ >= 23
      const i32x8 z8 = {0, 0, 0, 0, 0, 0, 0, 0};
      __builtin_amdgcn_tensor_load_to_lds(g0, g1, z4, z4, z8, 0);
#else
      __builtin_amdgcn_tensor_load_to_lds(g0, g1, z4, z4, 0);
#endif
      wait_tensorcnt0();
      staged = true;
    }
#endif
    if (!staged && lane < 20) {
      float4 v;
      if (elemBase + TILE_ELEMS <= (long)B) {
        v = ((const float4*)(in + elemBase * 10))[lane];
      } else {
        float t4[4];
        const long base = elemBase * 10;
#pragma unroll
        for (int j = 0; j < 4; ++j) {
          const long gi = base + lane * 4 + j;
          t4[j] = (gi < (long)B * 10) ? in[gi] : 0.f;
        }
        v = make_float4(t4[0], t4[1], t4[2], t4[3]);
      }
      ((float4*)IB)[lane] = v;
    }
    wave_lds_fence();

    // ---- embed + exact GELU -> X[16][32] (lane = feature, pads zeroed) ----
#pragma unroll
    for (int r = 0; r < 16; ++r) {
      float v = 0.0f;
      if (lane < E) {
        float acc = be;
#pragma unroll
        for (int j = 0; j < 5; ++j) acc += IB[r * 5 + j] * we[j];
        v = gelu_exact(acc);
      }
      X[r][lane] = v;
    }
    wave_lds_fence();

    // ---- QKV projection: [16x30] x [30x90] via 6 WMMAs ----
    {
      const v16h a = load_a_frag(&X[0][0], lane);
#pragma unroll
      for (int t = 0; t < 6; ++t) {
        const v8f c = wmma16(a, Bqkv[t]);
        const int col = t * 16 + n16;
        if (col < 90) {
#pragma unroll
          for (int r = 0; r < 8; ++r) Q[r + hh * 8][col] = c[r] + bqkv[t];
        }
      }
    }
    wave_lds_fence();

    // ---- attention: seq=2, 3 heads, D=10 (one lane per element x head) ----
    if (lane < 24) {
      const int e  = lane / 3;
      const int hd = lane % 3;
      const int r0 = 2 * e, r1 = r0 + 1;
      const int qo = hd * HD, ko = 30 + hd * HD, vo = 60 + hd * HD;
      float s00 = 0.f, s01 = 0.f, s10 = 0.f, s11 = 0.f;
#pragma unroll
      for (int d = 0; d < HD; ++d) {
        const float q0 = Q[r0][qo + d], q1 = Q[r1][qo + d];
        const float k0 = Q[r0][ko + d], k1 = Q[r1][ko + d];
        s00 += q0 * k0; s01 += q0 * k1;
        s10 += q1 * k0; s11 += q1 * k1;
      }
      const float scale = 0.31622776601683794f;  // 1/sqrt(10)
      s00 *= scale; s01 *= scale; s10 *= scale; s11 *= scale;
      const float m0 = fmaxf(s00, s01), m1 = fmaxf(s10, s11);
      const float e00 = __expf(s00 - m0), e01 = __expf(s01 - m0);
      const float e10 = __expf(s10 - m1), e11 = __expf(s11 - m1);
      const float i0 = 1.0f / (e00 + e01), i1 = 1.0f / (e10 + e11);
#pragma unroll
      for (int d = 0; d < HD; ++d) {
        const float v0 = Q[r0][vo + d], v1 = Q[r1][vo + d];
        C[r0][qo + d] = (e00 * v0 + e01 * v1) * i0;
        C[r1][qo + d] = (e10 * v0 + e11 * v1) * i1;
      }
    } else {                                   // lanes 24..31 zero C pad cols
      const int r = (lane - 24) * 2;
      C[r][30] = 0.f;     C[r][31] = 0.f;
      C[r + 1][30] = 0.f; C[r + 1][31] = 0.f;
    }
    wave_lds_fence();

    // ---- attention out-proj + residual -> X ----
    {
      const v16h a = load_a_frag(&C[0][0], lane);
#pragma unroll
      for (int t = 0; t < 2; ++t) {
        const v8f c = wmma16(a, Bo[t]);
        const int col = t * 16 + n16;
        if (col < E) {
#pragma unroll
          for (int r = 0; r < 8; ++r) {
            const int row = r + hh * 8;
            X[row][col] = c[r] + bo2[t] + X[row][col];
          }
        }
      }
    }
    wave_lds_fence();
    layernorm16(X, lane, gamma, beta);   // LN1
    wave_lds_fence();

    // ---- FFN fc1 + exact GELU -> C ----
    {
      const v16h a = load_a_frag(&X[0][0], lane);
#pragma unroll
      for (int t = 0; t < 2; ++t) {
        const v8f c = wmma16(a, B1f[t]);
        const int col = t * 16 + n16;
        if (col < E) {
#pragma unroll
          for (int r = 0; r < 8; ++r)
            C[r + hh * 8][col] = gelu_exact(c[r] + bb1[t]);
        }
      }
    }
    wave_lds_fence();

    // ---- FFN fc2 + residual -> X, then LN2 (same gamma/beta) ----
    {
      const v16h a = load_a_frag(&C[0][0], lane);
#pragma unroll
      for (int t = 0; t < 2; ++t) {
        const v8f c = wmma16(a, B2f[t]);
        const int col = t * 16 + n16;
        if (col < E) {
#pragma unroll
          for (int r = 0; r < 8; ++r) {
            const int row = r + hh * 8;
            X[row][col] = X[row][col] + c[r] + bb2[t];
          }
        }
      }
    }
    wave_lds_fence();
    layernorm16(X, lane, gamma, beta);   // LN2
    wave_lds_fence();

    // ---- head: logits of token 1 (one lane per element x output) ----
    if (lane < 24) {
      const int e = lane / 3, o = lane % 3;
      const long elem = elemBase + e;
      if (elem < B) {
        const float* row = X[2 * e + 1];
        float acc = b_o[o];
#pragma unroll
        for (int d = 0; d < E; ++d) acc += row[d] * w_o[o * E + d];
        out[elem * 3 + o] = acc;
      }
    }
    wave_lds_fence();  // X / IB reused next iteration
  }
}

extern "C" void kernel_launch(void* const* d_in, const int* in_sizes, int n_in,
                              void* d_out, int out_size, void* d_ws, size_t ws_size,
                              hipStream_t stream) {
  const float* in    = (const float*)d_in[0];
  const float* w_emb = (const float*)d_in[1];
  const float* b_emb = (const float*)d_in[2];
  const float* w_in  = (const float*)d_in[3];
  const float* b_in  = (const float*)d_in[4];
  const float* w_out = (const float*)d_in[5];
  const float* b_out = (const float*)d_in[6];
  const float* w1    = (const float*)d_in[7];
  const float* b1    = (const float*)d_in[8];
  const float* w2    = (const float*)d_in[9];
  const float* b2    = (const float*)d_in[10];
  const float* gamma = (const float*)d_in[11];
  const float* beta  = (const float*)d_in[12];
  const float* w_o   = (const float*)d_in[13];
  const float* b_o   = (const float*)d_in[14];
  float* out = (float*)d_out;

  const int B = in_sizes[0] / 10;                        // [B,2,5]
  const int elemsPerBlock = WAVES * TILE_ELEMS * NITER;  // 320
  const int blocks = (B + elemsPerBlock - 1) / elemsPerBlock;  // 3125 @ B=1e6

  tiny_transformer_wmma<<<blocks, WAVES * 32, 0, stream>>>(
      in, w_emb, b_emb, w_in, b_in, w_out, b_out,
      w1, b1, w2, b2, gamma, beta, w_o, b_o, out, B);
}
